// HetDTAGraph_77421080477770
// MI455X (gfx1250) — matrix operand order repaired and checked by reading, not compiled, not run on hardware
//
#include <hip/hip_runtime.h>
#include <hip/hip_bf16.h>
#include <math.h>

#define N_NODE 50000
#define HIDF   128
#define NEDGE  600000

typedef __attribute__((ext_vector_type(16))) __bf16 v16bf;
typedef __attribute__((ext_vector_type(8)))  float  v8f;

// ------------------------------ device helpers ------------------------------

__device__ __forceinline__ unsigned enc_mono(float f) {
    unsigned b = __float_as_uint(f);
    return (b & 0x80000000u) ? ~b : (b | 0x80000000u);
}
__device__ __forceinline__ float dec_mono(unsigned u) {
    unsigned b = (u & 0x80000000u) ? (u & 0x7FFFFFFFu) : ~u;
    return __uint_as_float(b);
}
__device__ __forceinline__ void edge_ids(int idx, int emain,
                                         const int* __restrict__ src,
                                         const int* __restrict__ dst,
                                         int& s, int& d) {
    if (idx < emain) { s = src[idx]; d = dst[idx]; }
    else             { s = idx - emain; d = s; }         // appended self loops
}
__device__ __forceinline__ float lrelu(float x) { return x > 0.f ? x : 0.2f * x; }

// ------------------------------ GEMM (WMMA) ---------------------------------
// C[M x 128] = act(A)[M x 128] @ W[128 x 128], bf16 MACs, f32 accumulate.
// One wave per 16-row strip; whole W^T staged in LDS as bf16 (32 KB).
__global__ __launch_bounds__(256)
void gat_gemm_wmma(const float* __restrict__ A, const float* __restrict__ W,
                   float* __restrict__ C, int M, int relu_in) {
    __shared__ __align__(32) __bf16 s_wT[HIDF * HIDF];   // [n][k]
    const int t = threadIdx.x;
    for (int idx = t; idx < HIDF * HIDF; idx += 256) {
        int k = idx >> 7, n = idx & 127;
        s_wT[n * HIDF + k] = (__bf16)W[idx];
    }
    __syncthreads();

    const int wave = t >> 5;
    const int lane = t & 31;
    const int m0 = (blockIdx.x * 8 + wave) * 16;
    if (m0 >= M) return;

    const int r   = lane & 15;        // row of A tile / col of B tile
    const int hi  = lane >> 4;        // lane half
    const int klo = hi * 16;          // A-fragment K sub-offset (contiguous K = 16*hi + i)

    // A fragments: 4 k-steps of 32. Element i<8 -> K = ks*32+klo+i,
    // element i>=8 -> K = ks*32+8+klo+(i-8)  (contiguous K = 16*hi + i, matches B).
    v16bf a_frag[4];
    const float* arow = A + (size_t)(m0 + r) * HIDF;
    #pragma unroll
    for (int ks = 0; ks < 4; ++ks) {
        const float* ap = arow + ks * 32;
        #pragma unroll
        for (int i = 0; i < 8; ++i) {
            float x0 = ap[klo + i];
            float x1 = ap[8 + klo + i];
            if (relu_in) { x0 = fmaxf(x0, 0.f); x1 = fmaxf(x1, 0.f); }
            a_frag[ks][i]     = (__bf16)x0;
            a_frag[ks][i + 8] = (__bf16)x1;
        }
    }

    #pragma unroll
    for (int nt = 0; nt < 8; ++nt) {
        const int n0 = nt * 16;
        v8f acc = {};
        #pragma unroll
        for (int ks = 0; ks < 4; ++ks) {
            // B (32x16), contiguous K order as A: lanes 0-15 hold K=0..15, lanes 16-31 K=16..31,
            // N = lane&15. K contiguous per lane -> one aligned 32B LDS read.
            const __bf16* bp = s_wT + (size_t)(n0 + r) * HIDF + ks * 32 + hi * 16;
            v16bf b_frag = *(const v16bf*)bp;
            acc = __builtin_amdgcn_wmma_f32_16x16x32_bf16(
                false, a_frag[ks], false, b_frag, (short)0, acc, false, false);
        }
        // C layout: VGPR v -> rows m0+v (lanes 0-15) and m0+v+8 (lanes 16-31)
        #pragma unroll
        for (int v = 0; v < 8; ++v)
            C[(size_t)(m0 + v + hi * 8) * HIDF + n0 + r] = acc[v];
    }
}

// ------------------------- per-row dot with attention vec --------------------
__global__ __launch_bounds__(256)
void gat_rowdot(const float* __restrict__ H, const float* __restrict__ a,
                float* __restrict__ e, int M) {
    int gid = blockIdx.x * blockDim.x + threadIdx.x;
    int row = gid >> 5, lane = gid & 31;
    if (row >= M) return;
    float4 x = ((const float4*)(H + (size_t)row * HIDF))[lane];
    float4 w = ((const float4*)a)[lane];
    float p = x.x * w.x + x.y * w.y + x.z * w.z + x.w * w.w;
    #pragma unroll
    for (int off = 16; off > 0; off >>= 1) p += __shfl_xor(p, off, 32);
    if (lane == 0) e[row] = p;
}

// ----------------------------- segment softmax -------------------------------
__global__ void gat_init_stats(unsigned* __restrict__ m_enc,
                               float* __restrict__ ssum, int n) {
    int i = blockIdx.x * blockDim.x + threadIdx.x;
    if (i >= n) return;
    m_enc[i] = enc_mono(-INFINITY);
    ssum[i]  = 0.f;
}

__global__ void gat_edge_max(const int* __restrict__ src, const int* __restrict__ dst,
                             const float* __restrict__ es, const float* __restrict__ ed,
                             unsigned* __restrict__ m_enc, int emain, int etot) {
    int idx = blockIdx.x * blockDim.x + threadIdx.x;
    if (idx >= etot) return;
    int s, d; edge_ids(idx, emain, src, dst, s, d);
    float v = lrelu(es[s] + ed[d]);
    atomicMax(m_enc + d, enc_mono(v));
}

__global__ void gat_fix_m(float* __restrict__ m, int n) {   // decode in place
    int i = blockIdx.x * blockDim.x + threadIdx.x;
    if (i >= n) return;
    float f = dec_mono(((unsigned*)m)[i]);
    m[i] = __builtin_isfinite(f) ? f : 0.f;
}

__global__ void gat_edge_sum(const int* __restrict__ src, const int* __restrict__ dst,
                             const float* __restrict__ es, const float* __restrict__ ed,
                             const float* __restrict__ m, float* __restrict__ ssum,
                             int emain, int etot) {
    int idx = blockIdx.x * blockDim.x + threadIdx.x;
    if (idx >= etot) return;
    int s, d; edge_ids(idx, emain, src, dst, s, d);
    float v = lrelu(es[s] + ed[d]);
    atomicAdd(ssum + d, __expf(v - m[d]));
}

__global__ void gat_recip_s(float* __restrict__ ssum, int n) {
    int i = blockIdx.x * blockDim.x + threadIdx.x;
    if (i >= n) return;
    ssum[i] = 1.f / (ssum[i] + 1e-16f);
}

// one wave per edge: gather 512B feature row, scatter 128 f32 atomic adds
__global__ __launch_bounds__(256)
void gat_edge_scatter(const int* __restrict__ src, const int* __restrict__ dst,
                      const float* __restrict__ es, const float* __restrict__ ed,
                      const float* __restrict__ m, const float* __restrict__ sinv,
                      const float* __restrict__ H, float* __restrict__ out,
                      int emain, int etot) {
    int gid  = blockIdx.x * blockDim.x + threadIdx.x;
    int eidx = gid >> 5, lane = gid & 31;
    if (eidx >= etot) return;
    int s, d; edge_ids(eidx, emain, src, dst, s, d);
    float v = lrelu(es[s] + ed[d]);
    float alpha = __expf(v - m[d]) * sinv[d];
    float4 x = ((const float4*)(H + (size_t)s * HIDF))[lane];
    float* op = out + (size_t)d * HIDF + lane * 4;
    atomicAdd(op + 0, alpha * x.x);
    atomicAdd(op + 1, alpha * x.y);
    atomicAdd(op + 2, alpha * x.z);
    atomicAdd(op + 3, alpha * x.w);
}

// ---------------------------- misc elementwise -------------------------------
__global__ void gat_init_out_bias(float* __restrict__ out, const float* __restrict__ ba,
                                  const float* __restrict__ bb, int total) {
    int i = blockIdx.x * blockDim.x + threadIdx.x;
    if (i >= total) return;
    int f = i & 127;
    out[i] = ba[f] + bb[f];
}

__global__ void gat_relu_inplace(float* __restrict__ p, int total) {
    int i = blockIdx.x * blockDim.x + threadIdx.x;
    if (i >= total) return;
    p[i] = fmaxf(p[i], 0.f);
}

// ------------------------------ host orchestration ---------------------------

static void run_relation(hipStream_t st,
                         const float* x_src, const float* x_dst, bool same_x, bool relu_in,
                         const float* W, const float* a_s, const float* a_d,
                         const int* ei, int n_src, int n_dst, bool self_loops,
                         float* h_src, float* h_dst,
                         float* e_src, float* e_dst, float* m, float* ssum,
                         float* out_accum) {
    const int* src = ei;
    const int* dst = ei + NEDGE;
    const dim3 blk(256);

    gat_gemm_wmma<<<(n_src + 127) / 128, blk, 0, st>>>(x_src, W, h_src, n_src, relu_in);
    const float* hd = h_src;
    if (!same_x) {
        gat_gemm_wmma<<<(n_dst + 127) / 128, blk, 0, st>>>(x_dst, W, h_dst, n_dst, relu_in);
        hd = h_dst;
    }
    gat_rowdot<<<(n_src * 32 + 255) / 256, blk, 0, st>>>(h_src, a_s, e_src, n_src);
    gat_rowdot<<<(n_dst * 32 + 255) / 256, blk, 0, st>>>(hd, a_d, e_dst, n_dst);

    const int etot = NEDGE + (self_loops ? n_dst : 0);
    gat_init_stats<<<(n_dst + 255) / 256, blk, 0, st>>>((unsigned*)m, ssum, n_dst);
    gat_edge_max<<<(etot + 255) / 256, blk, 0, st>>>(src, dst, e_src, e_dst,
                                                     (unsigned*)m, NEDGE, etot);
    gat_fix_m<<<(n_dst + 255) / 256, blk, 0, st>>>(m, n_dst);
    gat_edge_sum<<<(etot + 255) / 256, blk, 0, st>>>(src, dst, e_src, e_dst, m, ssum,
                                                     NEDGE, etot);
    gat_recip_s<<<(n_dst + 255) / 256, blk, 0, st>>>(ssum, n_dst);
    long long th = (long long)etot * 32;
    gat_edge_scatter<<<(int)((th + 255) / 256), blk, 0, st>>>(src, dst, e_src, e_dst,
                                                              m, ssum, h_src, out_accum,
                                                              NEDGE, etot);
}

static void run_layer(hipStream_t st, const float* xd, const float* xt, bool relu_in,
                      const float* W, const float* as, const float* ad, const float* b,
                      const int* ei_dd, const int* ei_dt, const int* ei_td, const int* ei_tt,
                      float* h1, float* h2, float* es, float* edv, float* m, float* ss,
                      float* out_d, float* out_t) {
    const int tot = N_NODE * HIDF;
    const dim3 blk(256);
    // HeteroConv sum init: out_d = b_dd + b_td ; out_t = b_dt + b_tt
    gat_init_out_bias<<<(tot + 255) / 256, blk, 0, st>>>(out_d, b + 0 * HIDF, b + 2 * HIDF, tot);
    gat_init_out_bias<<<(tot + 255) / 256, blk, 0, st>>>(out_t, b + 1 * HIDF, b + 3 * HIDF, tot);

    // 0: drug->drug (self loops)
    run_relation(st, xd, xd, true,  relu_in, W + 0 * HIDF * HIDF, as + 0 * HIDF, ad + 0 * HIDF,
                 ei_dd, N_NODE, N_NODE, true,  h1, h2, es, edv, m, ss, out_d);
    // 1: drug->target
    run_relation(st, xd, xt, false, relu_in, W + 1 * HIDF * HIDF, as + 1 * HIDF, ad + 1 * HIDF,
                 ei_dt, N_NODE, N_NODE, false, h1, h2, es, edv, m, ss, out_t);
    // 2: target->drug
    run_relation(st, xt, xd, false, relu_in, W + 2 * HIDF * HIDF, as + 2 * HIDF, ad + 2 * HIDF,
                 ei_td, N_NODE, N_NODE, false, h1, h2, es, edv, m, ss, out_d);
    // 3: target->target (self loops)
    run_relation(st, xt, xt, true,  relu_in, W + 3 * HIDF * HIDF, as + 3 * HIDF, ad + 3 * HIDF,
                 ei_tt, N_NODE, N_NODE, true,  h1, h2, es, edv, m, ss, out_t);
}

extern "C" void kernel_launch(void* const* d_in, const int* in_sizes, int n_in,
                              void* d_out, int out_size, void* d_ws, size_t ws_size,
                              hipStream_t stream) {
    const float* x_drug   = (const float*)d_in[0];
    const float* x_target = (const float*)d_in[1];
    const float* W1  = (const float*)d_in[2];
    const float* as1 = (const float*)d_in[3];
    const float* ad1 = (const float*)d_in[4];
    const float* b1  = (const float*)d_in[5];
    const float* W2  = (const float*)d_in[6];
    const float* as2 = (const float*)d_in[7];
    const float* ad2 = (const float*)d_in[8];
    const float* b2  = (const float*)d_in[9];
    const int* ei_dd = (const int*)d_in[10];
    const int* ei_dt = (const int*)d_in[11];
    const int* ei_td = (const int*)d_in[12];
    const int* ei_tt = (const int*)d_in[13];

    // workspace carve-out (all fully overwritten each call)
    float* p   = (float*)d_ws;
    float* h1  = p; p += (size_t)N_NODE * HIDF;   // projected src features
    float* h2  = p; p += (size_t)N_NODE * HIDF;   // projected dst features
    float* od1 = p; p += (size_t)N_NODE * HIDF;   // layer-1 drug output
    float* ot1 = p; p += (size_t)N_NODE * HIDF;   // layer-1 target output
    float* es  = p; p += N_NODE;                  // per-src logits
    float* edv = p; p += N_NODE;                  // per-dst logits
    float* m   = p; p += N_NODE;                  // segment max (uint-encoded then f32)
    float* ss  = p; p += N_NODE;                  // segment sum -> reciprocal

    float* out_d2 = (float*)d_out;                       // final drug out
    float* out_t2 = (float*)d_out + (size_t)N_NODE * HIDF;  // final target out

    // layer 1 (raw inputs), layer 2 (ReLU fused into GEMM input conversion)
    run_layer(stream, x_drug, x_target, false, W1, as1, ad1, b1,
              ei_dd, ei_dt, ei_td, ei_tt, h1, h2, es, edv, m, ss, od1, ot1);
    run_layer(stream, od1, ot1, true, W2, as2, ad2, b2,
              ei_dd, ei_dt, ei_td, ei_tt, h1, h2, es, edv, m, ss, out_d2, out_t2);

    // final ReLU in place on concatenated outputs
    const int tot = 2 * N_NODE * HIDF;
    gat_relu_inplace<<<(tot + 255) / 256, 256, 0, stream>>>((float*)d_out, tot);
}
